// Multihead_attention_88484916232619
// MI455X (gfx1250) — matrix operand order repaired and checked by reading, not compiled, not run on hardware
//
#include <hip/hip_runtime.h>

#define B_  16
#define T_  1024
#define D_  512
#define H_  8
#define DH_ 64

typedef __attribute__((ext_vector_type(16))) __bf16 v16bf;
typedef __attribute__((ext_vector_type(8)))  float  v8f;
typedef int v4i_b128 __attribute__((vector_size(16)));

#define GLOBAL_AS __attribute__((address_space(1)))
#define LDS_AS    __attribute__((address_space(3)))

union Frag16 {
  v16bf  v;
  uint4  u[2];
  __bf16 e[16];
};

static __device__ __forceinline__ v8f wmma_bf16(v16bf a, v16bf b, v8f c) {
  // D = A(16x32 bf16) * B(32x16 bf16) + C(16x16 f32)
  return __builtin_amdgcn_wmma_f32_16x16x32_bf16(false, a, false, b, (short)0, c,
                                                 false, false);
}

// 16-byte async global -> LDS copy (CDNA5 GLOBAL_LOAD_ASYNC_TO_LDS_B128)
static __device__ __forceinline__ void async_ld16(const void* g, void* l) {
#if __has_builtin(__builtin_amdgcn_global_load_async_to_lds_b128)
  __builtin_amdgcn_global_load_async_to_lds_b128(
      (GLOBAL_AS v4i_b128*)(GLOBAL_AS void*)(void*)g,
      (LDS_AS v4i_b128*)(LDS_AS void*)l, 0, 0);
#else
  unsigned loff = (unsigned)(unsigned long long)l;  // LDS aperture: low 32 bits
  asm volatile("global_load_async_to_lds_b128 %0, %1, off"
               :: "v"(loff), "v"(g) : "memory");
#endif
}

static __device__ __forceinline__ void wait_async0() {
#if __has_builtin(__builtin_amdgcn_s_wait_asynccnt)
  __builtin_amdgcn_s_wait_asynccnt(0);
#else
  asm volatile("s_wait_asynccnt 0x0" ::: "memory");
#endif
}

// ---------------------------------------------------------------------------
// Padding masks: mask[row] = sign(|sum_d X[row,d]|). One wave per row.
// ---------------------------------------------------------------------------
__global__ __launch_bounds__(256) void mask_kernel(const float* __restrict__ q,
                                                   const float* __restrict__ k,
                                                   float* __restrict__ qmask,
                                                   float* __restrict__ kmask) {
  int row  = blockIdx.x * 8 + (threadIdx.x >> 5);
  int lane = threadIdx.x & 31;
  if (row >= B_ * T_) return;
  const float* qp = q + (size_t)row * D_;
  const float* kp = k + (size_t)row * D_;
  float sq = 0.f, sk = 0.f;
  for (int d = lane; d < D_; d += 32) { sq += qp[d]; sk += kp[d]; }
#pragma unroll
  for (int off = 16; off > 0; off >>= 1) {
    sq += __shfl_xor(sq, off, 32);
    sk += __shfl_xor(sk, off, 32);
  }
  if (lane == 0) {
    qmask[row] = (sq != 0.f) ? 1.f : 0.f;
    kmask[row] = (sk != 0.f) ? 1.f : 0.f;
  }
}

// ---------------------------------------------------------------------------
// Weight convert + transpose: Wt[n][k] = (bf16) W[k][n]   (512x512)
// ---------------------------------------------------------------------------
__global__ __launch_bounds__(256) void wcvt_kernel(const float* __restrict__ W,
                                                   __bf16* __restrict__ Wt) {
  int n = blockIdx.x;
  for (int kk = threadIdx.x; kk < D_; kk += 256)
    Wt[(size_t)n * D_ + kk] = (__bf16)W[(size_t)kk * D_ + n];
}

// ---------------------------------------------------------------------------
// Projection GEMM: C[m,n] = (sum_k X[m,k] * Wt[n,k] + bias[n]) * oscale
// oscale = 1/sqrt(DH) for the Q projection (power of two -> exact in bf16),
// 1.0 otherwise.
// MODE 0/1: store bf16 row-major [m*D + n]  (Q, K)
// MODE 2  : store bf16 transposed per head Vt[((b*H+h)*DH+dh)*T + t]
// One wave computes one 16x16 tile; K=512 -> 16 WMMAs.
// ---------------------------------------------------------------------------
template <int MODE>
__global__ __launch_bounds__(256) void proj_kernel(const float* __restrict__ X,
                                                   const __bf16* __restrict__ Wt,
                                                   const float* __restrict__ bias,
                                                   __bf16* __restrict__ outp,
                                                   float oscale) {
  int wave = threadIdx.x >> 5;
  int lane = threadIdx.x & 31;
  int lm = lane & 15, lh = lane >> 4;
  int m0 = (blockIdx.x * 8 + wave) * 16;
  int n0 = blockIdx.y * 16;

  v8f c = {};
  const float*  xrow = X  + (size_t)(m0 + lm) * D_;
  const __bf16* wrow = Wt + (size_t)(n0 + lm) * D_;

  for (int k0 = 0; k0 < D_; k0 += 32) {
    Frag16 a;
    const float* xp = xrow + k0 + lh * 8;
    float4 x0 = *(const float4*)(xp);
    float4 x1 = *(const float4*)(xp + 4);
    float4 x2 = *(const float4*)(xp + 16);
    float4 x3 = *(const float4*)(xp + 20);
    a.e[0]=(__bf16)x0.x;  a.e[1]=(__bf16)x0.y;  a.e[2]=(__bf16)x0.z;  a.e[3]=(__bf16)x0.w;
    a.e[4]=(__bf16)x1.x;  a.e[5]=(__bf16)x1.y;  a.e[6]=(__bf16)x1.z;  a.e[7]=(__bf16)x1.w;
    a.e[8]=(__bf16)x2.x;  a.e[9]=(__bf16)x2.y;  a.e[10]=(__bf16)x2.z; a.e[11]=(__bf16)x2.w;
    a.e[12]=(__bf16)x3.x; a.e[13]=(__bf16)x3.y; a.e[14]=(__bf16)x3.z; a.e[15]=(__bf16)x3.w;
    Frag16 bw;
    bw.u[0] = *(const uint4*)(wrow + k0 + lh * 8);
    bw.u[1] = *(const uint4*)(wrow + k0 + 16 + lh * 8);
    v16bf bfrag = bw.v;
    c = wmma_bf16(a.v, bfrag, c);
  }

  float bc = bias[n0 + lm];
#pragma unroll
  for (int r = 0; r < 8; r++) {
    int m = m0 + r + 8 * lh;               // C layout: M = r + 8*half, N = lane%16
    float val = (c[r] + bc) * oscale;
    if (MODE < 2) {
      outp[(size_t)m * D_ + n0 + lm] = (__bf16)val;
    } else {
      int b = m / T_, t = m % T_;
      int ncol = n0 + lm;
      int h = ncol >> 6, dh = ncol & 63;
      outp[(((size_t)b * H_ + h) * DH_ + dh) * T_ + t] = (__bf16)val;
    }
  }
}

// ---------------------------------------------------------------------------
// Flash-style causal attention.
// Workgroup = 128 threads (4 waves) = 64 query rows of one (b,h).
// 32-key K/V tiles double-buffered in LDS via async global->LDS DMA;
// each wave runs the online-softmax WMMA pipeline on its 16 rows.
// Row sums of P are computed with an extra WMMA against an all-ones B tile
// (replaces 32 ds_bpermute reductions per key tile).
// ---------------------------------------------------------------------------
__global__ __launch_bounds__(128) void attn_kernel(const __bf16* __restrict__ Qbf,
                                                   const __bf16* __restrict__ Kbf,
                                                   const __bf16* __restrict__ Vt,
                                                   const float* __restrict__ kmask,
                                                   const float* __restrict__ qmask,
                                                   const float* __restrict__ queries,
                                                   float* __restrict__ outp) {
  __shared__ __align__(16) __bf16 kbuf[2][32 * 64];   // [key][feat]
  __shared__ __align__(16) __bf16 vbuf[2][64 * 32];   // [feat][key]
  __shared__ __align__(16) __bf16 pshare[4][16 * 32]; // per-wave P tile

  int tid  = threadIdx.x;
  int wave = tid >> 5;
  int lane = tid & 31;
  int lm = lane & 15, lh = lane >> 4;
  int q0blk = blockIdx.x * 64;
  int q0 = q0blk + wave * 16;            // this wave's query rows
  int bh = blockIdx.y;
  int b = bh >> 3;                       // H_ == 8
  int h = bh & 7;

  const __bf16* Kg = Kbf + (size_t)b * T_ * D_ + h * DH_;  // + key*D_ + feat
  const __bf16* Vg = Vt + (size_t)bh * DH_ * T_;           // + feat*T_ + key
  const int nt = (q0blk + 64) / 32;      // number of 32-key tiles (causal)

  // Issue async copy of tile t into buffer t&1 (whole block cooperates).
  auto issue = [&](int t) {
    int k0 = t * 32, buf = t & 1;
#pragma unroll
    for (int i = 0; i < 2; i++) {
      int idx = tid + i * 128;
      int key = idx >> 3, ch = idx & 7;                    // 32x8 chunks of 8 bf16
      async_ld16(Kg + (size_t)(k0 + key) * D_ + ch * 8,
                 &kbuf[buf][key * 64 + ch * 8]);
      int feat = idx >> 2, vch = idx & 3;                  // 64x4 chunks of 8 bf16
      async_ld16(Vg + (size_t)feat * T_ + k0 + vch * 8,
                 &vbuf[buf][feat * 32 + vch * 8]);
    }
  };

  issue(0);

  // Q A-fragments (feature dim split into K=0..31 and K=32..63); Q pre-scaled
  Frag16 aq[2];
  const __bf16* qrow = Qbf + ((size_t)b * T_ + q0 + lm) * D_ + h * DH_;
#pragma unroll
  for (int f = 0; f < 2; f++) {
    aq[f].u[0] = *(const uint4*)(qrow + f * 32 + lh * 8);
    aq[f].u[1] = *(const uint4*)(qrow + f * 32 + 16 + lh * 8);
  }

  // Constant all-ones B fragment for P row sums via WMMA
  Frag16 ones;
#pragma unroll
  for (int i = 0; i < 16; i++) ones.e[i] = (__bf16)1.0f;

  float mrow[8], lrow[8];
  v8f o[4];
  v8f vz = {};
#pragma unroll
  for (int i = 0; i < 4; i++) o[i] = vz;
#pragma unroll
  for (int r = 0; r < 8; r++) { mrow[r] = -3.0e38f; lrow[r] = 0.f; }

  const float NEGV = -4.2949673e9f;
  __bf16* ps = pshare[wave];

  for (int t = 0; t < nt; t++) {
    wait_async0();        // my tile-t chunks have landed in LDS
    __syncthreads();      // everyone's landed; everyone done reading other buffer
    if (t + 1 < nt) issue(t + 1);   // overlap DMA of t+1 with compute on t

    int k0 = t * 32;
    if (k0 <= q0 + 15) {  // wave-uniform: skip tiles fully above the causal edge
      const __bf16* kb = kbuf[t & 1];
      const __bf16* vb = vbuf[t & 1];

      v8f s[2];
      float tmax[8];
#pragma unroll
      for (int r = 0; r < 8; r++) tmax[r] = -3.0e38f;

#pragma unroll
      for (int nh = 0; nh < 2; nh++) {
        s[nh] = vz;
#pragma unroll
        for (int f = 0; f < 2; f++) {
          Frag16 bkf;
          bkf.u[0] = *(const uint4*)(kb + (nh * 16 + lm) * 64 + f * 32 + lh * 8);
          bkf.u[1] = *(const uint4*)(kb + (nh * 16 + lm) * 64 + f * 32 + 16 + lh * 8);
          v16bf bk = bkf.v;
          s[nh] = wmma_bf16(aq[f].v, bk, s[nh]);
        }
        float km = kmask[(size_t)b * T_ + k0 + nh * 16 + lm];
        int kglob = k0 + nh * 16 + lm;
#pragma unroll
        for (int r = 0; r < 8; r++) {
          float sv = s[nh][r];                       // already scaled by 1/sqrt(DH)
          int qr = q0 + r + 8 * lh;
          if (km == 0.f || kglob > qr) sv = NEGV;    // key-pad + causal masks
          s[nh][r] = sv;
          tmax[r] = fmaxf(tmax[r], sv);
        }
      }

      // Row max across the 16 lanes holding each row: step-outer loop keeps the
      // 8 independent bpermute chains in flight together (one DS wait per step).
#pragma unroll
      for (int off = 1; off < 16; off <<= 1) {
#pragma unroll
        for (int r = 0; r < 8; r++)
          tmax[r] = fmaxf(tmax[r], __shfl_xor(tmax[r], off, 16));
      }
      float alpha[8];
#pragma unroll
      for (int r = 0; r < 8; r++) {
        float mn = fmaxf(mrow[r], tmax[r]);
        alpha[r] = __expf(mrow[r] - mn);
        mrow[r] = mn;
      }
#pragma unroll
      for (int ct = 0; ct < 4; ct++)
#pragma unroll
        for (int r = 0; r < 8; r++) o[ct][r] *= alpha[r];

      // P = exp(S - m), staged (bf16) into this wave's LDS tile
#pragma unroll
      for (int nh = 0; nh < 2; nh++)
#pragma unroll
        for (int r = 0; r < 8; r++) {
          float p = __expf(s[nh][r] - mrow[r]);
          ps[(r + 8 * lh) * 32 + nh * 16 + lm] = (__bf16)p;
        }

      __threadfence_block();   // order within-wave LDS write -> cross-lane read
      // Relayout P: C layout (N in lanes) -> A layout (M in lanes, K in VGPRs)
      Frag16 ap;
      const __bf16* prow = ps + lm * 32 + lh * 8;
      ap.u[0] = *(const uint4*)(prow);
      ap.u[1] = *(const uint4*)(prow + 16);

      // Row sums of P via WMMA against all-ones B (replaces shuffle reduction)
      v8f srow = wmma_bf16(ap.v, ones.v, vz);
#pragma unroll
      for (int r = 0; r < 8; r++) lrow[r] = lrow[r] * alpha[r] + srow[r];

      // O += P * V over 4 feature column-tiles
#pragma unroll
      for (int ct = 0; ct < 4; ct++) {
        Frag16 bvf;
        bvf.u[0] = *(const uint4*)(vb + (ct * 16 + lm) * 32 + lh * 8);
        bvf.u[1] = *(const uint4*)(vb + (ct * 16 + lm) * 32 + 16 + lh * 8);
        v16bf bv = bvf.v;
        o[ct] = wmma_bf16(ap.v, bv, o[ct]);
      }
    }
  }

  // Epilogue: normalize, query-pad mask (post-softmax), residual, fp32 store
#pragma unroll
  for (int r = 0; r < 8; r++) {
    int qr = q0 + r + 8 * lh;
    float qm = qmask[(size_t)b * T_ + qr];
    float inv = (lrow[r] > 0.f) ? (qm / lrow[r]) : 0.f;
#pragma unroll
    for (int ct = 0; ct < 4; ct++) {
      size_t idx = ((size_t)b * T_ + qr) * D_ + h * DH_ + ct * 16 + lm;
      outp[idx] = o[ct][r] * inv + queries[idx];
    }
  }
}

// ---------------------------------------------------------------------------
extern "C" void kernel_launch(void* const* d_in, const int* in_sizes, int n_in,
                              void* d_out, int out_size, void* d_ws, size_t ws_size,
                              hipStream_t stream) {
  (void)in_sizes; (void)n_in; (void)out_size; (void)ws_size;
  const float* queries = (const float*)d_in[0];
  const float* keys    = (const float*)d_in[1];
  const float* value   = (const float*)d_in[2];
  const float* Wq = (const float*)d_in[3];
  const float* bq = (const float*)d_in[4];
  const float* Wk = (const float*)d_in[5];
  const float* bk = (const float*)d_in[6];
  const float* Wv = (const float*)d_in[7];
  const float* bv = (const float*)d_in[8];
  float* out = (float*)d_out;

  char* ws = (char*)d_ws;
  const size_t QKB = (size_t)B_ * T_ * D_ * 2;   // bf16 activation tensor bytes
  const size_t WB  = (size_t)D_ * D_ * 2;        // bf16 weight bytes
  __bf16* Qbf = (__bf16*)(ws);
  __bf16* Kbf = (__bf16*)(ws + QKB);
  __bf16* Vt  = (__bf16*)(ws + 2 * QKB);
  __bf16* Wtq = (__bf16*)(ws + 3 * QKB);
  __bf16* Wtk = (__bf16*)(ws + 3 * QKB + WB);
  __bf16* Wtv = (__bf16*)(ws + 3 * QKB + 2 * WB);
  float* km   = (float*)(ws + 3 * QKB + 3 * WB);
  float* qm   = km + (size_t)B_ * T_;

  mask_kernel<<<dim3((B_ * T_) / 8), dim3(256), 0, stream>>>(queries, keys, qm, km);
  wcvt_kernel<<<dim3(D_), dim3(256), 0, stream>>>(Wq, Wtq);
  wcvt_kernel<<<dim3(D_), dim3(256), 0, stream>>>(Wk, Wtk);
  wcvt_kernel<<<dim3(D_), dim3(256), 0, stream>>>(Wv, Wtv);

  dim3 pgrid((B_ * T_) / 128, D_ / 16);          // 8 waves/block, wave = 16x16 tile
  proj_kernel<0><<<pgrid, dim3(256), 0, stream>>>(queries, Wtq, bq, Qbf, 0.125f);
  proj_kernel<1><<<pgrid, dim3(256), 0, stream>>>(keys,    Wtk, bk, Kbf, 1.0f);
  proj_kernel<2><<<pgrid, dim3(256), 0, stream>>>(value,   Wtv, bv, Vt,  1.0f);

  attn_kernel<<<dim3(T_ / 64, B_ * H_), dim3(128), 0, stream>>>(Qbf, Kbf, Vt, km, qm,
                                                                queries, out);
}
